// Correlation_3968549782067
// MI455X (gfx1250) — compile-verified, matrix operand order checked by reading, not run on hardware
//
#include <hip/hip_runtime.h>
#include <hip/hip_bf16.h>
#include <cstdint>
#include <cstddef>

// ---------- CDNA5 types ----------
typedef __attribute__((ext_vector_type(16))) __bf16       v16bf;
typedef __attribute__((ext_vector_type(8)))  __bf16       v8bf;
typedef __attribute__((ext_vector_type(8)))  float        v8f;
typedef __attribute__((ext_vector_type(4)))  unsigned int uint32x4;
typedef __attribute__((ext_vector_type(8)))  int          int32x8;
typedef __attribute__((ext_vector_type(4)))  int          int32x4;

// ---------- problem geometry ----------
#define BS   32
#define CIN  256
#define QH   17
#define QW   17
#define H    65
#define W    65
#define KTOT (CIN * QW)     // 4352 : k = dx*256 + c   (dx-major packing)
#define KST  (KTOT / 32)    // 136 WMMA K-steps of 32
#define NT   2              // dy tiles: dy in [0,32), valid [0,17)
#define SW   96             // padded x rows (x+dx in [0,96))
#define SROW 264            // LDS row stride in bf16 (132 dwords == 4 mod 64 banks)
#define CCH  32             // channels staged per TDM transfer
#define NPIX (BS * H * W)   // 135200

// ---------- LDS layout (bytes) ----------
#define SBF_BYTES  ((size_t)SW * SROW * 2)   // 50688 : sBf[x8][c] bf16
#define STAGE_OFF  SBF_BYTES                 // f32 TDM staging after it

// ---------- ws layout (bytes) ----------
// Qbf: [b][ntile][kstep][lane(32)][elem(16)] __bf16 : B-matrix fragments
#define QBF_ELEMS ((size_t)BS * NT * KST * 32 * 16)      // 4,456,448 bf16
#define CORR_OFF  (QBF_ELEMS * 2)                        // 8,912,896 B
#define STATS_OFF (CORR_OFF + (size_t)NPIX * 4)

// 16-bit A/B fragment K mapping (ISA 7.12.2): element e of lane-group g:
//   e in [0,8)  -> j = e + 8*g          (contiguous run of 8)
//   e in [8,16) -> j = e + 8 + 8*g      (contiguous run of 8)
__device__ __forceinline__ int kmap(int e, int g) {
    return (e < 8) ? (e + 8 * g) : (e + 8 + 8 * g);
}

// ---------- k0: zero corr + stats ----------
__global__ void zero_ws_kernel(float* p, int n) {
    int i = blockIdx.x * 256 + threadIdx.x;
    if (i < n) p[i] = 0.0f;
}

// ---------- k1: build bf16 B-matrix fragments from query ----------
// B(k = dx*256+c, n = dy) = Q[b, c, dy, dx]; zero for dy >= 17.
__global__ void prep_q_kernel(const float* __restrict__ q, __bf16* __restrict__ qbf) {
    const int kk = blockIdx.x;   // 0..135
    const int nt = blockIdx.y;   // 0..1
    const int b  = blockIdx.z;   // 0..31
    for (int s = threadIdx.x; s < 512; s += 256) {
        const int lane = s >> 4;
        const int e    = s & 15;
        const int g    = lane >> 4;
        const int j    = kmap(e, g);
        const int k    = kk * 32 + j;           // < 4352
        const int dx   = k >> 8;                // 0..16
        const int c    = k & 255;
        const int n    = nt * 16 + (lane & 15); // dy
        float val = (n < QH) ? q[(((size_t)b * CIN + c) * QH + n) * QW + dx] : 0.0f;
        size_t idx = ((((size_t)(b * NT + nt) * KST + kk) * 32 + lane) * 16) + e;
        qbf[idx] = (__bf16)val;
    }
}

// ---------- k2: WMMA correlation row kernel ----------
// One workgroup per (sy, b). 10 waves: wave w -> (mtile = w>>1, ntile = w&1).
// T(x,dy) = sum_{c,dx} S[b,c,sy,x+dx-8] * Q[b,c,dy,dx]; scattered into corr
// along the dy-diagonal: corr[sy-dy+8, x] += T(x,dy).
__global__ __launch_bounds__(320) void corr_wmma_kernel(
        const float* __restrict__ smap,
        const __bf16* __restrict__ qbf,
        float* __restrict__ corr) {
    extern __shared__ unsigned char smem[];
    __bf16* sBf   = (__bf16*)smem;               // sBf[x8][c], row stride SROW
    float*  stage = (float*)(smem + STAGE_OFF);  // CCH*W f32 TDM staging

    const int sy   = blockIdx.x;
    const int b    = blockIdx.y;
    const int tid  = threadIdx.x;
    const int lane = tid & 31;
    const int wid  = tid >> 5;

    // zero the padded A buffer (x halo + row-stride pad stay zero)
    for (int i = tid; i < SW * SROW; i += 320) sBf[i] = (__bf16)0.0f;
    __syncthreads();

    const float* srow0 = smap + (size_t)b * CIN * (H * W) + (size_t)sy * W;

    for (int cc = 0; cc < CIN / CCH; ++cc) {
        const float* gsrc = srow0 + (size_t)cc * CCH * (H * W);
#if __has_builtin(__builtin_amdgcn_tensor_load_to_lds)
        // TDM: async-load a 2D f32 tile (CCH rows of 65 cols, stride 4225) to LDS.
        // Only wave 0 issues (TDM ignores EXEC; scalar branch keeps others away).
        if (__builtin_amdgcn_readfirstlane((int)wid) == 0) {
            const uint64_t ga = (uint64_t)(uintptr_t)gsrc;
            uint32x4 g0;
            g0.x = 1u;                                   // count=1, user descriptor
            g0.y = (unsigned)STAGE_OFF;                  // lds_addr
            g0.z = (uint32_t)ga;                         // global_addr[31:0]
            g0.w = (uint32_t)((ga >> 32) & 0x01FFFFFFu)  // global_addr[56:32]
                   | 0x80000000u;                        // type=2 ("image")
            int32x8 g1;
            g1[0] = (int)(2u << 16);                     // data_size = 4 bytes
            g1[1] = (int)(((unsigned)W & 0xFFFFu) << 16);   // tensor_dim0 = 65
            g1[2] = (int)(((unsigned)CCH & 0xFFFFu) << 16); // tensor_dim1 = 32
            g1[3] = (int)((unsigned)W << 16);            // tile_dim0 = 65
            g1[4] = CCH;                                 // tile_dim1 = 32
            g1[5] = H * W;                               // tensor_dim0_stride = 4225
            g1[6] = 0;
            g1[7] = 0;
            int32x4 gz = (int32x4)0;
#if __clang_major__ >= 23
            __builtin_amdgcn_tensor_load_to_lds(g0, g1, gz, gz, (int32x8)0, 0);
#else
            __builtin_amdgcn_tensor_load_to_lds(g0, g1, gz, gz, 0);
#endif
#if __has_builtin(__builtin_amdgcn_s_wait_tensorcnt)
            __builtin_amdgcn_s_wait_tensorcnt(0);
#else
#warning "CDNA5: s_wait_tensorcnt builtin missing, using inline asm"
            asm volatile("s_wait_tensorcnt 0x0" ::: "memory");
#endif
        }
#else
#warning "CDNA5: tensor_load_to_lds builtin NOT available, using plain-copy fallback"
        for (int i = tid; i < CCH * W; i += 320)
            stage[i] = gsrc[(size_t)(i / W) * (H * W) + (i % W)];
#endif
        __syncthreads();
        // f32 -> bf16 transposed into A layout: sBf[x+8][c] = S[b,c,sy,x]
        for (int i = tid; i < CCH * W; i += 320) {
            int c = i / W, x = i - c * W;
            sBf[(x + 8) * SROW + (cc * CCH + c)] = (__bf16)stage[i];
        }
        __syncthreads();
    }

    // ---- WMMA main loop ----
    const int mt = wid >> 1;           // 0..4
    const int nt = wid & 1;            // 0..1
    const int M  = lane & 15;
    const int g  = lane >> 4;
    const int xg = mt * 16 + M;        // A-row = x position (0..79)

    const __bf16* qb = qbf + (((size_t)(b * NT + nt) * KST) * 32 + lane) * 16;

    v8f acc = {};
#pragma unroll 4
    for (int kk = 0; kk < KST; ++kk) {
        // B fragment: 32 contiguous bytes from precomputed global layout (L2-hot)
        v16bf bfrag = *(const v16bf*)(qb + (size_t)kk * 32 * 16);
        // A fragment: two contiguous 8-channel runs -> two ds_load_b128
        int k0  = kk * 32 + 8 * g;         // run 1: elements 0..7
        int k1  = k0 + 16;                 // run 2: elements 8..15
        int dx0 = k0 >> 8, c0 = k0 & 255;
        int dx1 = k1 >> 8, c1 = k1 & 255;
        v8bf lo = *(const v8bf*)&sBf[(xg + dx0) * SROW + c0];
        v8bf hi = *(const v8bf*)&sBf[(xg + dx1) * SROW + c1];
        v16bf afrag = __builtin_shufflevector(lo, hi,
                0, 1, 2, 3, 4, 5, 6, 7, 8, 9, 10, 11, 12, 13, 14, 15);
        acc = __builtin_amdgcn_wmma_f32_16x16x32_bf16(
                  false, afrag, false, bfrag, (short)0, acc, false, false);
    }

    // ---- diagonal scatter epilogue: corr[sy-dy+8, x] += T(x, dy) ----
    const int dy = nt * 16 + M;        // D-layout: N = lane&15
#pragma unroll
    for (int r = 0; r < 8; ++r) {
        int x = mt * 16 + r + 8 * g;   // D-layout: M = vgpr + 8*(lane>>4)
        int y = sy - dy + 8;
        if (dy < QH && x < W && y >= 0 && y < H)
            atomicAdd(&corr[((size_t)b * H + y) * W + x], acc[r]);
    }
}

// ---------- k3: global sum / sumsq ----------
__global__ void stats_kernel(const float* __restrict__ corr, float* __restrict__ stats) {
    __shared__ float s0[256], s1[256];
    float a = 0.0f, b = 0.0f;
    for (int i = blockIdx.x * 256 + threadIdx.x; i < NPIX; i += gridDim.x * 256) {
        float v = corr[i];
        a += v;
        b += v * v;
    }
    s0[threadIdx.x] = a;
    s1[threadIdx.x] = b;
    __syncthreads();
    for (int st = 128; st > 0; st >>= 1) {
        if (threadIdx.x < st) {
            s0[threadIdx.x] += s0[threadIdx.x + st];
            s1[threadIdx.x] += s1[threadIdx.x + st];
        }
        __syncthreads();
    }
    if (threadIdx.x == 0) {
        atomicAdd(&stats[0], s0[0]);
        atomicAdd(&stats[1], s1[0]);
    }
}

// ---------- k4: batchnorm finalize ----------
__global__ void bn_kernel(const float* __restrict__ corr, const float* __restrict__ stats,
                          const float* __restrict__ gamma, const float* __restrict__ beta,
                          float* __restrict__ out) {
    int i = blockIdx.x * 256 + threadIdx.x;
    if (i >= NPIX) return;
    const float inv_n = 1.0f / (float)NPIX;
    float mean = stats[0] * inv_n;
    float var  = stats[1] * inv_n - mean * mean;
    float scale = rsqrtf(var + 1e-5f) * gamma[0];
    out[i] = (corr[i] - mean) * scale + beta[0];
}

// ---------- host launch ----------
extern "C" void kernel_launch(void* const* d_in, const int* in_sizes, int n_in,
                              void* d_out, int out_size, void* d_ws, size_t ws_size,
                              hipStream_t stream) {
    (void)in_sizes; (void)n_in; (void)out_size; (void)ws_size;
    const float* q     = (const float*)d_in[0];
    const float* smap  = (const float*)d_in[1];
    const float* gamma = (const float*)d_in[2];
    const float* beta  = (const float*)d_in[3];
    float* out = (float*)d_out;

    __bf16* qbf  = (__bf16*)d_ws;
    float* corr  = (float*)((char*)d_ws + CORR_OFF);
    float* stats = (float*)((char*)d_ws + STATS_OFF);

    // k0: zero corr (+ the 2 stats words right after it)
    zero_ws_kernel<<<(NPIX + 2 + 255) / 256, 256, 0, stream>>>(corr, NPIX + 2);

    // k1: query -> bf16 B fragments
    prep_q_kernel<<<dim3(KST, NT, BS), 256, 0, stream>>>(q, qbf);

    // k2: WMMA correlation (one WG per (row, batch)); ~50KB A + 8.3KB TDM stage
    size_t shbytes = SBF_BYTES + (size_t)CCH * W * 4;  // 59,008 B
    corr_wmma_kernel<<<dim3(H, BS), 320, shbytes, stream>>>(smap, qbf, corr);

    // k3/k4: batchnorm statistics + normalization
    stats_kernel<<<256, 256, 0, stream>>>(corr, stats);
    bn_kernel<<<(NPIX + 255) / 256, 256, 0, stream>>>(corr, stats, gamma, beta, out);
}
